// GeometricAttention_1563368096336
// MI455X (gfx1250) — matrix operand order repaired and checked by reading, not compiled, not run on hardware
//
#include <hip/hip_runtime.h>

// ---- CDNA5 WMMA types -------------------------------------------------------
typedef __attribute__((ext_vector_type(16))) __bf16 v16bf;
typedef __attribute__((ext_vector_type(8)))  float  v8f;

#define B_    4
#define H_    8
#define N_    2048
#define DMV   128        // mv part of head dim (8*16)
#define DS_   16         // s part of head dim
#define D_    144        // full head dim
#define KSTR  168        // sQ/sK row stride (elems): 160 used + pad, 336B rows (16B aligned)
#define VSTR  72         // sVt row stride (elems): 64 keys + pad, 144B rows (16B aligned)
#define BM    64         // query rows per block (4 waves * 16)
#define BN    64         // key block per iteration (2 WMMA K-steps for P*V)
#define NW    4          // waves per block
#define LOG2E 1.44269504088896f

// _IPF = [1,1,-1,-1,-1,-1,-1,-1,1,1,1,1,1,1,-1,-1]  -> negative-bit mask
#define IPF_MASK 0xC0FCu

// truncating f32 -> bf16 (1 VALU)
__device__ __forceinline__ unsigned short f2bf(float f) {
  union { float f; unsigned int u; } c; c.f = f;
  return (unsigned short)(c.u >> 16);
}

// two f32 -> packed 2x bf16 via one V_PERM_B32: result = {hi16(b), hi16(a)}
__device__ __forceinline__ unsigned int pack2bf(float a, float b) {
  union { float f; unsigned int u; } x, y; x.f = a; y.f = b;
  return __builtin_amdgcn_perm(y.u, x.u, 0x07060302u);
}

// 16 bf16 operand from two 16B LDS chunks (two ds_load_b128)
__device__ __forceinline__ v16bf ld16(const unsigned short* p0,
                                      const unsigned short* p1) {
  union { v16bf v; uint4 u[2]; } r;
  r.u[0] = *(const uint4*)p0;
  r.u[1] = *(const uint4*)p1;
  return r.v;
}

#define WMMA_BF16(A, B, C) \
  __builtin_amdgcn_wmma_f32_16x16x32_bf16(false, (A), false, (B), (short)0, (C), false, false)

// store a (4 d-values) x (2 keys) patch into transposed sVt as packed b32s
__device__ __forceinline__ void vt_store4(unsigned short* sVt, int d, int kp,
                                          float4 a, float4 b) {
  *(unsigned int*)&sVt[(d + 0) * VSTR + 2 * kp] = pack2bf(a.x, b.x);
  *(unsigned int*)&sVt[(d + 1) * VSTR + 2 * kp] = pack2bf(a.y, b.y);
  *(unsigned int*)&sVt[(d + 2) * VSTR + 2 * kp] = pack2bf(a.z, b.z);
  *(unsigned int*)&sVt[(d + 3) * VSTR + 2 * kp] = pack2bf(a.w, b.w);
}

__global__ void __launch_bounds__(128)
__attribute__((amdgpu_waves_per_eu(2)))
geo_attn_kernel(const float* __restrict__ q_mv, const float* __restrict__ k_mv,
                const float* __restrict__ v_mv, const float* __restrict__ q_s,
                const float* __restrict__ k_s,  const float* __restrict__ v_s,
                float* __restrict__ out)
{
  __shared__ __align__(16) unsigned short sQ [BM * KSTR];     // bf16 Q, d padded
  __shared__ __align__(16) unsigned short sK [BN * KSTR];     // bf16 K, d padded
  __shared__ __align__(16) unsigned short sVt[160 * VSTR];    // bf16 V^T (+ ones row d=144)
  __shared__ __align__(16) unsigned short sP [NW * 16 * BN];  // per-wave P scratch

  const int tid  = threadIdx.x;
  const int wave = tid >> 5;
  const int lane = tid & 31;
  const int bh   = blockIdx.y;
  const int m0   = blockIdx.x * BM;

  const float scale = 1.0f / 12.0f;        // 1/sqrt(144), folded into Q

  const float* qmvb = q_mv + (size_t)bh * N_ * DMV;
  const float* qsb  = q_s  + (size_t)bh * N_ * DS_;
  const float* kmvb = k_mv + (size_t)bh * N_ * DMV;
  const float* ksb  = k_s  + (size_t)bh * N_ * DS_;
  const float* vmvb = v_mv + (size_t)bh * N_ * DMV;
  const float* vsb  = v_s  + (size_t)bh * N_ * DS_;

  // ---- one-time LDS init ---------------------------------------------------
  for (int idx = tid; idx < BM * 16; idx += 128) {            // zero pad d=144..159
    int r = idx >> 4, c = idx & 15;
    sQ[r * KSTR + 144 + c] = 0;
    sK[r * KSTR + 144 + c] = 0;
  }
  for (int idx = tid; idx < 16 * VSTR; idx += 128) {          // ones row d=144, zeros after
    sVt[144 * VSTR + idx] = (idx < VSTR) ? (unsigned short)0x3F80 : (unsigned short)0;
  }

  // ---- stage Q once: sign pattern + scale, packed bf16 stores --------------
  {
    int r  = tid >> 1;
    int cb = (tid & 1) * 64;
    const float* src = qmvb + (size_t)(m0 + r) * DMV + cb;
    unsigned short* dst = &sQ[r * KSTR + cb];
    #pragma unroll
    for (int i = 0; i < 16; ++i) {
      float4 q = *(const float4*)(src + 4 * i);
      int c = cb + 4 * i;
      float f0 = ((IPF_MASK >> ((c + 0) & 15)) & 1u) ? -scale : scale;
      float f1 = ((IPF_MASK >> ((c + 1) & 15)) & 1u) ? -scale : scale;
      float f2 = ((IPF_MASK >> ((c + 2) & 15)) & 1u) ? -scale : scale;
      float f3 = ((IPF_MASK >> ((c + 3) & 15)) & 1u) ? -scale : scale;
      uint2 u; u.x = pack2bf(q.x * f0, q.y * f1); u.y = pack2bf(q.z * f2, q.w * f3);
      *(uint2*)(dst + 4 * i) = u;
    }
    const float* src2 = qsb + (size_t)(m0 + r) * DS_ + (tid & 1) * 8;
    float4 a = *(const float4*)src2;
    float4 b = *(const float4*)(src2 + 4);
    uint4 u; u.x = pack2bf(a.x * scale, a.y * scale); u.y = pack2bf(a.z * scale, a.w * scale);
    u.z = pack2bf(b.x * scale, b.y * scale); u.w = pack2bf(b.z * scale, b.w * scale);
    *(uint4*)&sQ[r * KSTR + 128 + (tid & 1) * 8] = u;
  }
  __syncthreads();

  // ---- WMMA lane geometry (A/B: contiguous k = 16*half + s) ------------------------------------
  const int arow  = lane & 15;
  const int ac0   = (lane < 16) ? 0 : 16;  // A: lane half -> K 0..15 vs 16..31
  const int bklo  = (lane < 16) ? 0 : 16;  // B: lane half -> K 0..15 vs 16..31
  const int rbase = (lane >> 4) * 8;       // C/D: lane half -> rows 0..7 vs 8..15
  const int cidx  = lane & 15;
  const unsigned short* qrow = &sQ[(wave * 16 + arow) * KSTR];

  v8f acc[10] = {};                        // tiles 0..8 = O, tile 9 = row-sum (ones col)
  float mrow[8];
  #pragma unroll
  for (int v = 0; v < 8; ++v) mrow[v] = -3.0e38f;

  for (int j0 = 0; j0 < N_; j0 += BN) {
    __syncthreads();
    // ---- stage K: rows j0..j0+63, packed bf16 -------------------------
    {
      int r  = tid >> 1;
      int cb = (tid & 1) * 64;
      const float* src = kmvb + (size_t)(j0 + r) * DMV + cb;
      unsigned short* dst = &sK[r * KSTR + cb];
      #pragma unroll
      for (int i = 0; i < 16; ++i) {
        float4 q = *(const float4*)(src + 4 * i);
        uint2 u; u.x = pack2bf(q.x, q.y); u.y = pack2bf(q.z, q.w);
        *(uint2*)(dst + 4 * i) = u;
      }
      const float* src2 = ksb + (size_t)(j0 + r) * DS_ + (tid & 1) * 8;
      float4 a = *(const float4*)src2;
      float4 b = *(const float4*)(src2 + 4);
      uint4 u; u.x = pack2bf(a.x, a.y); u.y = pack2bf(a.z, a.w);
      u.z = pack2bf(b.x, b.y); u.w = pack2bf(b.z, b.w);
      *(uint4*)&sK[r * KSTR + 128 + (tid & 1) * 8] = u;
    }
    // ---- stage V transposed: key-pairs -> packed b32 stores -----------
    {
      int kp    = tid & 31;                // keys 2kp, 2kp+1
      int dbase = (tid >> 5) * 36;         // d range [dbase, dbase+36)
      const float* r0 = vmvb + (size_t)(j0 + 2 * kp) * DMV;
      const float* r1 = vmvb + (size_t)(j0 + 2 * kp + 1) * DMV;
      if (dbase != 108) {                  // pure mv groups
        #pragma unroll
        for (int i = 0; i < 9; ++i) {
          int d = dbase + 4 * i;
          vt_store4(sVt, d, kp, *(const float4*)(r0 + d), *(const float4*)(r1 + d));
        }
      } else {                             // mv 108..127 then s 128..143
        #pragma unroll
        for (int i = 0; i < 5; ++i) {
          int d = 108 + 4 * i;
          vt_store4(sVt, d, kp, *(const float4*)(r0 + d), *(const float4*)(r1 + d));
        }
        const float* s0p = vsb + (size_t)(j0 + 2 * kp) * DS_;
        const float* s1p = vsb + (size_t)(j0 + 2 * kp + 1) * DS_;
        #pragma unroll
        for (int i = 0; i < 4; ++i) {
          vt_store4(sVt, DMV + 4 * i, kp,
                    *(const float4*)(s0p + 4 * i), *(const float4*)(s1p + 4 * i));
        }
      }
    }
    __syncthreads();

    // prefetch next key block into L2 (global_prefetch_b8)
    if (j0 + BN < N_) {
      __builtin_prefetch(kmvb + (size_t)(j0 + BN) * DMV + tid * 64, 0, 0);
      __builtin_prefetch(vmvb + (size_t)(j0 + BN) * DMV + tid * 64, 0, 0);
    }

    // ---- S = Q * K^T : four 16x16 tiles, software-pipelined operands -------
    v8f s0 = {}, s1 = {}, s2 = {}, s3 = {};
    v16bf aC, b0C, b1C, b2C, b3C;
    {
      const unsigned short* qp = qrow + ac0;
      const unsigned short* k0 = &sK[(arow     ) * KSTR + bklo];
      const unsigned short* k1 = &sK[(arow + 16) * KSTR + bklo];
      const unsigned short* k2 = &sK[(arow + 32) * KSTR + bklo];
      const unsigned short* k3 = &sK[(arow + 48) * KSTR + bklo];
      aC  = ld16(qp, qp + 8);
      b0C = ld16(k0, k0 + 8);
      b1C = ld16(k1, k1 + 8);
      b2C = ld16(k2, k2 + 8);
      b3C = ld16(k3, k3 + 8);
    }
    #pragma unroll
    for (int d = 0; d < 5; ++d) {
      v16bf aN, b0N, b1N, b2N, b3N;
      if (d < 4) {      // issue next chunk's loads before this chunk's WMMAs
        const unsigned short* qp = qrow + (d + 1) * 32 + ac0;
        const unsigned short* k0 = &sK[(arow     ) * KSTR + (d + 1) * 32 + bklo];
        const unsigned short* k1 = &sK[(arow + 16) * KSTR + (d + 1) * 32 + bklo];
        const unsigned short* k2 = &sK[(arow + 32) * KSTR + (d + 1) * 32 + bklo];
        const unsigned short* k3 = &sK[(arow + 48) * KSTR + (d + 1) * 32 + bklo];
        aN  = ld16(qp, qp + 8);
        b0N = ld16(k0, k0 + 8);
        b1N = ld16(k1, k1 + 8);
        b2N = ld16(k2, k2 + 8);
        b3N = ld16(k3, k3 + 8);
      }
      s0 = WMMA_BF16(aC, b0C, s0);
      s1 = WMMA_BF16(aC, b1C, s1);
      s2 = WMMA_BF16(aC, b2C, s2);
      s3 = WMMA_BF16(aC, b3C, s3);
      if (d < 4) { aC = aN; b0C = b0N; b1C = b1N; b2C = b2N; b3C = b3N; }
    }

    // ---- online softmax: only the row-max needs cross-lane reduction --
    float mnew[8], alpha[8];
    #pragma unroll
    for (int v = 0; v < 8; ++v) {
      float mx = fmaxf(fmaxf(s0[v], s1[v]), fmaxf(s2[v], s3[v]));
      mx = fmaxf(mx, __shfl_xor(mx, 1, 32));
      mx = fmaxf(mx, __shfl_xor(mx, 2, 32));
      mx = fmaxf(mx, __shfl_xor(mx, 4, 32));
      mx = fmaxf(mx, __shfl_xor(mx, 8, 32));
      mnew[v]  = fmaxf(mrow[v], mx);
      alpha[v] = exp2f((mrow[v] - mnew[v]) * LOG2E);
      mrow[v]  = mnew[v];
    }
    // P = exp(S - m); write bf16 to per-wave scratch (C-layout -> A-layout)
    unsigned short* pw = &sP[wave * 16 * BN];
    #pragma unroll
    for (int v = 0; v < 8; ++v) {
      float e0 = exp2f((s0[v] - mnew[v]) * LOG2E);
      float e1 = exp2f((s1[v] - mnew[v]) * LOG2E);
      float e2 = exp2f((s2[v] - mnew[v]) * LOG2E);
      float e3 = exp2f((s3[v] - mnew[v]) * LOG2E);
      int ro = (rbase + v) * BN;
      pw[ro + cidx]      = f2bf(e0);
      pw[ro + 16 + cidx] = f2bf(e1);
      pw[ro + 32 + cidx] = f2bf(e2);
      pw[ro + 48 + cidx] = f2bf(e3);
    }
    // rescale accumulators (includes row-sum tile 9 -> denominator recurrence)
    #pragma unroll
    for (int t = 0; t < 10; ++t)
      #pragma unroll
      for (int v = 0; v < 8; ++v) acc[t][v] *= alpha[v];

    // same-wave DS ops are in-order; reload P as two A operands
    const unsigned short* pr = &pw[arow * BN];
    v16bf pa0 = ld16(pr + ac0,      pr + ac0 + 8);
    v16bf pa1 = ld16(pr + 32 + ac0, pr + 32 + ac0 + 8);

    // ---- acc += P * V : 10 tiles, software-pipelined B operands -------
    v16bf c0, c1;
    {
      const unsigned short* vb = &sVt[cidx * VSTR];
      c0 = ld16(vb + bklo,      vb + bklo + 8);
      c1 = ld16(vb + 32 + bklo, vb + 32 + bklo + 8);
    }
    #pragma unroll
    for (int t = 0; t < 10; ++t) {
      v16bf n0, n1;
      if (t < 9) {
        const unsigned short* vb = &sVt[((t + 1) * 16 + cidx) * VSTR];
        n0 = ld16(vb + bklo,      vb + bklo + 8);
        n1 = ld16(vb + 32 + bklo, vb + 32 + bklo + 8);
      }
      acc[t] = WMMA_BF16(pa0, c0, acc[t]);
      acc[t] = WMMA_BF16(pa1, c1, acc[t]);
      if (t < 9) { c0 = n0; c1 = n1; }
    }
  }

  // ---- epilogue: denominators live in acc[9] col 0; broadcast per half -----
  float rl[8];
  #pragma unroll
  for (int v = 0; v < 8; ++v) {
    float sum = __shfl(acc[9][v], lane & 16, 32);   // col 0 holder of this half
    rl[v] = 1.0f / sum;
  }

  float* out_mv = out;                                   // (B,H,N,8,16) flat
  float* out_s  = out + (size_t)B_ * H_ * N_ * DMV;      // (B,H,N,16) flat
  #pragma unroll
  for (int t = 0; t < 9; ++t) {
    int col = t * 16 + cidx;
    #pragma unroll
    for (int v = 0; v < 8; ++v) {
      int   row = m0 + wave * 16 + rbase + v;
      float val = acc[t][v] * rl[v];
      if (col < DMV)
        out_mv[(size_t)bh * N_ * DMV + (size_t)row * DMV + col] = val;
      else
        out_s [(size_t)bh * N_ * DS_ + (size_t)row * DS_ + (col - DMV)] = val;
    }
  }
}

extern "C" void kernel_launch(void* const* d_in, const int* in_sizes, int n_in,
                              void* d_out, int out_size, void* d_ws, size_t ws_size,
                              hipStream_t stream) {
  const float* q_mv = (const float*)d_in[0];
  const float* k_mv = (const float*)d_in[1];
  const float* v_mv = (const float*)d_in[2];
  const float* q_s  = (const float*)d_in[3];
  const float* k_s  = (const float*)d_in[4];
  const float* v_s  = (const float*)d_in[5];
  float* outp = (float*)d_out;

  dim3 grid(N_ / BM, B_ * H_, 1);   // 32 x 32 workgroups
  dim3 block(128, 1, 1);            // 4 wave32 waves
  geo_attn_kernel<<<grid, block, 0, stream>>>(q_mv, k_mv, v_mv, q_s, k_s, v_s, outp);
}
